// Model_89275190214911
// MI455X (gfx1250) — matrix operand order repaired and checked by reading, not compiled, not run on hardware
//
#include <hip/hip_runtime.h>
#include <hip/hip_bf16.h>

#define CN 1536
#define CS 48
#define GS 32
#define HID 256
#define NB 8
#define NT 8
#define TEXTD 300
#define CATSZ 716
#define OUTSZ 1536
#define NEGF  -1.0e9f
#define NINF  -3.0e38f

typedef __attribute__((ext_vector_type(16))) __bf16 v16bf;
typedef __attribute__((ext_vector_type(8)))  __bf16 v8bf;
typedef __attribute__((ext_vector_type(8)))  float  v8f;

struct Ptrs {
  // inputs
  const float *code_x,*divided,*neighbors,*cate,*text;
  const int   *lens,*user;
  const float *c_emb,*n_emb,*u_emb,*cate_emb,*adj,*cate_adj,*Wg,*bg,*Wc,*bc,
              *Wi,*bi,*Wh,*bh,*Wq,*Wk,*Wv,*Wd,*bd,*ctx,*ug_g,*ug_a,*ug_c,
              *gam,*bet,*Wcls,*bcls;
  // workspace (bf16)
  __bf16 *adjbf,*WgT,*WiT,*WhT,*WqT,*WkT,*wvT,*cadjbf;
  __bf16 *ceT,*neT,*tmp2T,*Xc,*Xn,*cobf,*xbf,*Qh,*Kh,*VTm,*hbf;
  // workspace (f32 / int)
  float *m1v,*m2v,*m23v,*no_f,*noprev,*h_f,*hm1,*outs,*pooled,*feat,*yb;
  unsigned *uom1,*uom23;
  int *anyf;
  float *out;
};

// ---------------- helpers ----------------
__device__ __forceinline__ v8f zero8(){ v8f z;
#pragma unroll
  for(int i=0;i<8;i++) z[i]=0.f; return z; }

__device__ __forceinline__ float lrelu(float x){ return x>0.f ? x : 0.01f*x; }

__device__ __forceinline__ unsigned f2ord(float f){
  unsigned u = __float_as_uint(f);
  return (u & 0x80000000u) ? ~u : (u | 0x80000000u);
}
__device__ __forceinline__ float ord2f(unsigned u){
  return (u & 0x80000000u) ? __uint_as_float(u & 0x7FFFFFFFu) : __uint_as_float(~u);
}

__device__ __forceinline__ v8f wmma_bf(v16bf a, v16bf b, v8f c){
  return __builtin_amdgcn_wmma_f32_16x16x32_bf16(false, a, false, b, (short)0, c, false, false);
}

// A fragment: 16x32 bf16, source row-major [row][K], ld in elements.
// ISA 7.12.2: lanes 0-15 hold K 0..7 & 16..23, lanes 16-31 hold K 8..15 & 24..31.
__device__ __forceinline__ v16bf frag_a(const __bf16* base, int ld, int m0, int k0){
  int lane = threadIdx.x & 31;
  int m = m0 + (lane & 15);
  int hi = lane >> 4;
  const __bf16* p = base + (size_t)m*ld + k0 + 8*hi;
  v8bf lo = *(const v8bf*)p;
  v8bf hh = *(const v8bf*)(p + 16);
  v16bf r;
#pragma unroll
  for(int i=0;i<8;i++){ r[i]=lo[i]; r[i+8]=hh[i]; }
  return r;
}

// B fragment: 32x16 bf16, source is B^T row-major [N][K].
// lanes 0-15 hold K 0..7 & 16..23, lanes 16-31 hold K 8..15 & 24..31.
__device__ __forceinline__ v16bf frag_b(const __bf16* baseT, int ld, int n0, int k0){
  int lane = threadIdx.x & 31;
  int n = n0 + (lane & 15);
  int hi = lane >> 4;
  const __bf16* p = baseT + (size_t)n*ld + k0 + 8*hi;
  v8bf lo = *(const v8bf*)p;
  v8bf hh = *(const v8bf*)(p + 16);
  v16bf r;
#pragma unroll
  for(int i=0;i<8;i++){ r[i]=lo[i]; r[i+8]=hh[i]; }
  return r;
}

// A fragment straight from a 32-float row (converted to bf16 on the fly).
__device__ __forceinline__ v16bf frag_a_f32row(const float* rowp){
  int lane = threadIdx.x & 31; int hi = lane >> 4;
  v16bf r;
#pragma unroll
  for(int i=0;i<8;i++){ r[i]=(__bf16)rowp[8*hi+i]; r[i+8]=(__bf16)rowp[16+8*hi+i]; }
  return r;
}

// ---------------- one-time prep: bf16 conversions + zero carried state ----------------
__global__ void k_init(Ptrs p){
  long long idx = (long long)blockIdx.x*blockDim.x + threadIdx.x;
  if (idx < (long long)CN*CN){ p.adjbf[idx] = (__bf16)p.adj[idx]; return; }
  idx -= (long long)CN*CN;
  if (idx < 32*64){ int n=idx/64, k=idx%64;
    p.WgT[idx] = (k<CS) ? (__bf16)p.Wg[k*GS+n] : (__bf16)0.f; return; }
  idx -= 32*64;
  if (idx < 768*32){ int n=idx/32, k=idx%32; p.WiT[idx]=(__bf16)p.Wi[k*768+n]; return; }
  idx -= 768*32;
  if (idx < 768*256){ int n=idx/256, k=idx%256; p.WhT[idx]=(__bf16)p.Wh[k*768+n]; return; }
  idx -= 768*256;
  if (idx < 32*32){ int n=idx/32, k=idx%32;
    p.WqT[idx]=(__bf16)p.Wq[k*32+n]; p.WkT[idx]=(__bf16)p.Wk[k*32+n]; return; }
  idx -= 32*32;
  if (idx < 256*32){ int n=idx/32, k=idx%32; p.wvT[idx]=(__bf16)p.Wv[k*256+n]; return; }
  idx -= 256*32;
  if (idx < CN*32){ p.cadjbf[idx]=(__bf16)p.cate_adj[idx]; return; }
  idx -= CN*32;
  if (idx < (long long)NB*CN*HID){ p.h_f[idx]=0.f; p.hbf[idx]=(__bf16)0.f; return; }
  idx -= (long long)NB*CN*HID;
  if (idx < NB*CN*32){ p.noprev[idx]=0.f; return; }
  idx -= NB*CN*32;
  if (idx < (long long)NB*CN*64){ p.Xc[idx]=(__bf16)0.f; p.Xn[idx]=(__bf16)0.f; return; }
}

// ---------------- per-step prep ----------------
__global__ void k_sprep(Ptrs p, int t){
  long long idx = (long long)blockIdx.x*blockDim.x + threadIdx.x;
  if (idx < (long long)NB*CS*CN){
    int b = (int)(idx/(CS*CN)); int r = (int)(idx%(CS*CN));
    int n = r/CN; int k = r%CN;
    float cm = p.code_x  [((size_t)b*NT+t)*CN + k];
    float nm = p.neighbors[((size_t)b*NT+t)*CN + k];
    p.ceT[(size_t)b*CS*CN + (size_t)n*CN + k] = (__bf16)(cm * p.c_emb[(size_t)k*CS+n]);
    p.neT[(size_t)b*CS*CN + (size_t)n*CN + k] = (__bf16)(nm * p.n_emb[(size_t)k*CS+n]);
    return;
  }
  idx -= (long long)NB*CS*CN;
  if (idx < NB*CN){
    int b = (int)(idx/CN), k = (int)(idx%CN);
    const float* d = p.divided + (((size_t)b*NT+t)*CN + k)*3;
    float m1=d[0], m2=d[1], m3=d[2];
    p.m1v[idx]=m1; p.m2v[idx]=m2;
    float m23=m2+m3; m23 = m23>1.f?1.f:(m23<0.f?0.f:m23);
    p.m23v[idx]=m23; return;
  }
  idx -= NB*CN;
  if (idx < NB*32*32){
    int b=(int)(idx/(32*32)); int r=(int)(idx%(32*32));
    int j=r/32, i=r%32;
    float cv = p.cate[((size_t)b*NT+t)*32 + i];
    float s = 0.f;
    if (cv > 0.f){
      for(int k=0;k<CS;k++) s += p.cate_emb[i*CS+k]*p.Wc[k*GS+j];
      s *= cv;
    }
    p.tmp2T[(size_t)b*32*32 + j*32 + i] = (__bf16)s; return;
  }
  idx -= NB*32*32;
  if (idx < NB*HID){ p.uom1[idx]=f2ord(NINF); p.uom23[idx]=f2ord(NINF); }
}

// ---------------- adj GEMM (ping-pong pipelined): a_ce/a_ne + X epilogue ----------------
__global__ void __launch_bounds__(32) k_adj(Ptrs p, int t){
  int b = blockIdx.x / 96; int m0 = (blockIdx.x % 96)*16;
  const __bf16* ceT = p.ceT + (size_t)b*CS*CN;
  const __bf16* neT = p.neT + (size_t)b*CS*CN;
  v8f ac[3], an[3];
#pragma unroll
  for(int j=0;j<3;j++){ ac[j]=zero8(); an[j]=zero8(); }

  // ping-pong buffers: loads write directly into alternating register sets,
  // no cur<-nxt copies, WMMA consumes the other buffer while loads fly.
  v16bf aA, bcA[3], bnA[3];
  v16bf aB, bcB[3], bnB[3];
  aA = frag_a(p.adjbf, CN, m0, 0);
#pragma unroll
  for(int j=0;j<3;j++){ bcA[j]=frag_b(ceT,CN,16*j,0); bnA[j]=frag_b(neT,CN,16*j,0); }

  for(int kb=0; kb<CN-64; kb+=64){
    aB = frag_a(p.adjbf, CN, m0, kb+32);
#pragma unroll
    for(int j=0;j<3;j++){ bcB[j]=frag_b(ceT,CN,16*j,kb+32); bnB[j]=frag_b(neT,CN,16*j,kb+32); }
#pragma unroll
    for(int j=0;j<3;j++){
      ac[j] = wmma_bf(aA, bcA[j], ac[j]);
      an[j] = wmma_bf(aA, bnA[j], an[j]);
    }
    aA = frag_a(p.adjbf, CN, m0, kb+64);
#pragma unroll
    for(int j=0;j<3;j++){ bcA[j]=frag_b(ceT,CN,16*j,kb+64); bnA[j]=frag_b(neT,CN,16*j,kb+64); }
#pragma unroll
    for(int j=0;j<3;j++){
      ac[j] = wmma_bf(aB, bcB[j], ac[j]);
      an[j] = wmma_bf(aB, bnB[j], an[j]);
    }
  }
  // tail: chunks CN-64 (in A) and CN-32 (load into B)
  aB = frag_a(p.adjbf, CN, m0, CN-32);
#pragma unroll
  for(int j=0;j<3;j++){ bcB[j]=frag_b(ceT,CN,16*j,CN-32); bnB[j]=frag_b(neT,CN,16*j,CN-32); }
#pragma unroll
  for(int j=0;j<3;j++){
    ac[j] = wmma_bf(aA, bcA[j], ac[j]);
    an[j] = wmma_bf(aA, bnA[j], an[j]);
  }
#pragma unroll
  for(int j=0;j<3;j++){
    ac[j] = wmma_bf(aB, bcB[j], ac[j]);
    an[j] = wmma_bf(aB, bnB[j], an[j]);
  }

  int lane=threadIdx.x&31, hi=lane>>4, nn=lane&15;
  const float* cx = p.code_x   + ((size_t)b*NT+t)*CN;
  const float* nb = p.neighbors+ ((size_t)b*NT+t)*CN;
  __bf16* Xc = p.Xc + (size_t)b*CN*64;
  __bf16* Xn = p.Xn + (size_t)b*CN*64;
#pragma unroll
  for(int j=0;j<3;j++){
    int col = 16*j + nn;
#pragma unroll
    for(int r=0;r<8;r++){
      int m = m0 + r + 8*hi;
      float cm = cx[m], nm = nb[m];
      float cev = cm * p.c_emb[(size_t)m*CS+col];
      float nev = nm * p.n_emb[(size_t)m*CS+col];
      float sum = ac[j][r] + an[j][r];
      Xc[(size_t)m*64+col] = (__bf16)(cev + cm*sum);
      Xn[(size_t)m*64+col] = (__bf16)(nev + nm*sum);
    }
  }
}

// ---------------- co/no = lrelu(X@Wg+bg), cao = lrelu(cate_adj@tmp2+bc), x = co+cao ----------------
__global__ void __launch_bounds__(32) k_co(Ptrs p){
  int b = blockIdx.x/96; int m0 = (blockIdx.x%96)*16;
  const __bf16* Xc = p.Xc + (size_t)b*CN*64;
  const __bf16* Xn = p.Xn + (size_t)b*CN*64;
  const __bf16* t2 = p.tmp2T + (size_t)b*32*32;
  v8f co[2],no[2],ca[2];
#pragma unroll
  for(int j=0;j<2;j++){ co[j]=zero8(); no[j]=zero8(); ca[j]=zero8(); }
  for(int kb=0; kb<64; kb+=32){
    v16bf axc = frag_a(Xc,64,m0,kb);
    v16bf axn = frag_a(Xn,64,m0,kb);
#pragma unroll
    for(int j=0;j<2;j++){
      v16bf bw = frag_b(p.WgT,64,16*j,kb);
      co[j]=wmma_bf(axc,bw,co[j]);
      no[j]=wmma_bf(axn,bw,no[j]);
    }
  }
  v16bf aca = frag_a(p.cadjbf,32,m0,0);
#pragma unroll
  for(int j=0;j<2;j++) ca[j]=wmma_bf(aca, frag_b(t2,32,16*j,0), ca[j]);

  int lane=threadIdx.x&31, hi=lane>>4, nn=lane&15;
  __bf16* cobf = p.cobf + (size_t)b*CN*32;
  __bf16* xbf  = p.xbf  + (size_t)b*CN*32;
  float*  nof  = p.no_f + (size_t)b*CN*32;
#pragma unroll
  for(int j=0;j<2;j++){
    int col=16*j+nn;
    float bgv=p.bg[col], bcv=p.bc[col];
#pragma unroll
    for(int r=0;r<8;r++){
      int m=m0+r+8*hi;
      float cov = lrelu(co[j][r]+bgv);
      float nov = lrelu(no[j][r]+bgv);
      float cav = lrelu(ca[j][r]+bcv);
      cobf[(size_t)m*32+col]=(__bf16)cov;
      nof [(size_t)m*32+col]=nov;
      xbf [(size_t)m*32+col]=(__bf16)(cov+cav);
    }
  }
}

// ---------------- Qh = (m2?no_prev:u_emb)@Wq * 1/sqrt(32), Kh = co@Wk ----------------
__global__ void __launch_bounds__(32) k_qk(Ptrs p){
  int b=blockIdx.x/96; int m0=(blockIdx.x%96)*16;
  int lane=threadIdx.x&31, hi=lane>>4, nn=lane&15;
  int m = m0 + (lane&15);
  const float* qrow = (p.m2v[(size_t)b*CN+m] > 0.f)
      ? (p.noprev + ((size_t)b*CN + m)*32)
      : (p.u_emb + (size_t)m*32);
  v16bf qa = frag_a_f32row(qrow);
  const __bf16* cobf = p.cobf + (size_t)b*CN*32;
  v16bf ca = frag_a(cobf,32,m0,0);
  __bf16* Qh = p.Qh + (size_t)b*CN*32;
  __bf16* Kh = p.Kh + (size_t)b*CN*32;
  const float scl = 0.17677669529663687f; // 1/sqrt(T_ATT)
#pragma unroll
  for(int j=0;j<2;j++){
    v8f aq = wmma_bf(qa, frag_b(p.WqT,32,16*j,0), zero8());
    v8f ak = wmma_bf(ca, frag_b(p.WkT,32,16*j,0), zero8());
    int col=16*j+nn;
#pragma unroll
    for(int r=0;r<8;r++){
      int mm=m0+r+8*hi;
      Qh[(size_t)mm*32+col]=(__bf16)(aq[r]*scl);
      Kh[(size_t)mm*32+col]=(__bf16)ak[r];
    }
  }
}

// ---------------- VT[c][row] = (co@Wv)^T (ping-pong over 96 column tiles) ----------------
__global__ void __launch_bounds__(32) k_vt(Ptrs p){
  int b=blockIdx.x/16; int m0=(blockIdx.x%16)*16;
  v16bf a = frag_a(p.wvT,32,m0,0);
  const __bf16* cobf = p.cobf + (size_t)b*CN*32;
  __bf16* VT = p.VTm + (size_t)b*HID*CN;
  int lane=threadIdx.x&31, hi=lane>>4, nn=lane&15;
  v16bf bA = frag_b(cobf,32,0,0), bB;
  for(int nt=0; nt<94; nt+=2){
    bB = frag_b(cobf,32,(nt+1)*16,0);
    v8f acc = wmma_bf(a, bA, zero8());
    int n = nt*16+nn;
#pragma unroll
    for(int r=0;r<8;r++) VT[(size_t)(m0+r+8*hi)*CN + n] = (__bf16)acc[r];
    bA = frag_b(cobf,32,(nt+2)*16,0);
    acc = wmma_bf(a, bB, zero8());
    n = (nt+1)*16+nn;
#pragma unroll
    for(int r=0;r<8;r++) VT[(size_t)(m0+r+8*hi)*CN + n] = (__bf16)acc[r];
  }
  bB = frag_b(cobf,32,95*16,0);
  {
    v8f acc = wmma_bf(a, bA, zero8());
    int n = 94*16+nn;
#pragma unroll
    for(int r=0;r<8;r++) VT[(size_t)(m0+r+8*hi)*CN + n] = (__bf16)acc[r];
    acc = wmma_bf(a, bB, zero8());
    n = 95*16+nn;
#pragma unroll
    for(int r=0;r<8;r++) VT[(size_t)(m0+r+8*hi)*CN + n] = (__bf16)acc[r];
  }
}

// ---------------- fused GRU: gi = x@Wi, gh = h@Wh (constant-trip, fully unrolled) ----------------
__global__ void __launch_bounds__(32) k_gru(Ptrs p){
  int b  = blockIdx.x/(96*16);
  int rm = blockIdx.x%(96*16);
  int m0 = (rm/16)*16; int n0 = (rm%16)*16;
  const __bf16* xbf = p.xbf + (size_t)b*CN*32;
  const __bf16* hbf = p.hbf + (size_t)b*CN*HID;
  v8f gi[3], gh[3];
#pragma unroll
  for(int g=0;g<3;g++){ gi[g]=zero8(); gh[g]=zero8(); }
  v16bf ax = frag_a(xbf,32,m0,0);
#pragma unroll
  for(int g=0;g<3;g++) gi[g]=wmma_bf(ax, frag_b(p.WiT,32,n0+256*g,0), gi[g]);
#pragma unroll
  for(int kb=0; kb<HID; kb+=32){
    v16bf ah = frag_a(hbf,HID,m0,kb);
#pragma unroll
    for(int g=0;g<3;g++) gh[g]=wmma_bf(ah, frag_b(p.WhT,HID,n0+256*g,kb), gh[g]);
  }

  int lane=threadIdx.x&31, hi=lane>>4, nn=lane&15;
  int col=n0+nn;
  float bi0=p.bi[col], bi1=p.bi[col+256], bi2=p.bi[col+512];
  float bh0=p.bh[col], bh1=p.bh[col+256], bh2=p.bh[col+512];
  const float* hf = p.h_f + (size_t)b*CN*HID;
  float* hm1 = p.hm1 + (size_t)b*CN*HID;
  const float* m1 = p.m1v + (size_t)b*CN;
  unsigned* uo = p.uom1 + (size_t)b*HID;
#pragma unroll
  for(int r=0;r<8;r++){
    int m=m0+r+8*hi;
    float hold = hf[(size_t)m*HID+col];
    float rr = 1.f/(1.f+__expf(-((gi[0][r]+bi0)+(gh[0][r]+bh0))));
    float zz = 1.f/(1.f+__expf(-((gi[1][r]+bi1)+(gh[1][r]+bh1))));
    float ng = tanhf((gi[2][r]+bi2) + rr*(gh[2][r]+bh2));
    float hm = (1.f-zz)*ng + zz*hold;
    hm1[(size_t)m*HID+col]=hm;
    if (m1[m]>0.f) atomicMax(&uo[col], f2ord(hm));
  }
}

// ---------------- flash attention: S = Qh@Kh^T (masked), online softmax, O = P@V ----------------
__global__ void __launch_bounds__(32) k_flash(Ptrs p){
  __shared__ __align__(16) __bf16 stage[16*32];
  int b=blockIdx.x/96; int m0=(blockIdx.x%96)*16;
  const __bf16* Qh = p.Qh  + (size_t)b*CN*32;
  const __bf16* Kh = p.Kh  + (size_t)b*CN*32;
  const __bf16* VT = p.VTm + (size_t)b*HID*CN;
  const float* m23 = p.m23v + (size_t)b*CN;
  int lane=threadIdx.x&31, hi=lane>>4, nn=lane&15;
  v16bf qa = frag_a(Qh,32,m0,0);
  v8f O[16];
#pragma unroll
  for(int j=0;j<16;j++) O[j]=zero8();
  float mcur[8], lcur[8];
#pragma unroll
  for(int r=0;r<8;r++){ mcur[r]=NINF; lcur[r]=0.f; }

  // K fragments for the first tile; next tile's fragments are issued
  // before the softmax/PV work of the current tile (branch-free clamp).
  v16bf kf0 = frag_b(Kh,32,0, 0);
  v16bf kf1 = frag_b(Kh,32,16,0);

  for(int k0=0; k0<CN; k0+=32){
    v8f s0 = wmma_bf(qa, kf0, zero8());
    v8f s1 = wmma_bf(qa, kf1, zero8());
    int kpre = (k0+32 < CN) ? (k0+32) : 0;   // clamped (redundant reload on last iter)
    kf0 = frag_b(Kh,32,kpre,   0);
    kf1 = frag_b(Kh,32,kpre+16,0);
    if (k0+64 < CN) __builtin_prefetch(Kh + (size_t)(k0+64+nn)*32, 0, 1);

    float msk0 = m23[k0+nn], msk1 = m23[k0+16+nn];
    float tmx[8];
#pragma unroll
    for(int r=0;r<8;r++){
      float a0 = msk0>0.f ? s0[r] : NINF;
      float a1 = msk1>0.f ? s1[r] : NINF;
      s0[r]=a0; s1[r]=a1;
      tmx[r]=fmaxf(a0,a1);
    }
#pragma unroll
    for(int d=1;d<16;d<<=1){
#pragma unroll
      for(int r=0;r<8;r++) tmx[r]=fmaxf(tmx[r], __shfl_xor(tmx[r], d, 32));
    }
    float alpha[8], rs[8];
#pragma unroll
    for(int r=0;r<8;r++){
      float mn = fmaxf(mcur[r], tmx[r]);
      float al = __expf(mcur[r]-mn);
      float p0 = (mn <= -1e37f) ? 0.f : __expf(s0[r]-mn);
      float p1 = (mn <= -1e37f) ? 0.f : __expf(s1[r]-mn);
      mcur[r]=mn; alpha[r]=al; s0[r]=p0; s1[r]=p1; rs[r]=p0+p1;
    }
#pragma unroll
    for(int d=1;d<16;d<<=1){
#pragma unroll
      for(int r=0;r<8;r++) rs[r] += __shfl_xor(rs[r], d, 32);
    }
#pragma unroll
    for(int r=0;r<8;r++) lcur[r]=lcur[r]*alpha[r]+rs[r];
    // stage P (16x32) row-major in LDS, then re-read in A-fragment layout
#pragma unroll
    for(int r=0;r<8;r++){
      stage[(r+8*hi)*32 + nn]      = (__bf16)s0[r];
      stage[(r+8*hi)*32 + 16 + nn] = (__bf16)s1[r];
    }
    asm volatile("s_wait_dscnt 0" ::: "memory");
    v16bf pa = frag_a(stage,32,0,0);
    // pipelined P@V (fully unrolled -> SSA, no register copies)
    v16bf vb_cur = frag_b(VT,CN,0,k0);
#pragma unroll
    for(int j=0;j<16;j++){
      v16bf vb_nxt;
      if (j < 15) vb_nxt = frag_b(VT,CN,16*(j+1),k0);
#pragma unroll
      for(int r=0;r<8;r++) O[j][r]*=alpha[r];
      O[j] = wmma_bf(pa, vb_cur, O[j]);
      vb_cur = vb_nxt;
    }
  }
  unsigned* uo = p.uom23 + (size_t)b*HID;
#pragma unroll
  for(int r=0;r<8;r++){
    int m=m0+r+8*hi;
    if (m23[m] > 0.f){
      float inv = (lcur[r]>0.f) ? 1.f/lcur[r] : 0.f;
#pragma unroll
      for(int j=0;j<16;j++){
        float hv = tanhf(O[j][r]*inv);
        atomicMax(&uo[16*j+nn], f2ord(hv));
      }
    }
  }
}

// ---------------- mask any-flags ----------------
__global__ void k_any(Ptrs p){
  int b = blockIdx.x;
  __shared__ int s1, s2;
  if (threadIdx.x==0){ s1=0; s2=0; }
  __syncthreads();
  int l1=0,l2=0;
  for(int k=threadIdx.x;k<CN;k+=blockDim.x){
    if (p.m1v [(size_t)b*CN+k]>0.f) l1=1;
    if (p.m23v[(size_t)b*CN+k]>0.f) l2=1;
  }
  if(l1) atomicOr(&s1,1);
  if(l2) atomicOr(&s2,1);
  __syncthreads();
  if (threadIdx.x==0){ p.anyf[b*2]=s1; p.anyf[b*2+1]=s2; }
}

// ---------------- carry update (gated by t < lens[b]) + outs[t] ----------------
__global__ void k_carry(Ptrs p, int t){
  long long idx = (long long)blockIdx.x*blockDim.x + threadIdx.x;
  if (idx < (long long)NB*CN*HID){
    int b = (int)(idx/(CN*HID));
    if (t < p.lens[b]){
      long long r = idx%(CN*HID); int m=(int)(r/HID);
      float hn = (p.m1v[(size_t)b*CN+m]>0.f) ? p.hm1[idx] : 0.f;
      p.h_f[idx]=hn; p.hbf[idx]=(__bf16)hn;
    }
    return;
  }
  idx -= (long long)NB*CN*HID;
  if (idx < NB*CN*32){
    int b=(int)(idx/(CN*32));
    if (t < p.lens[b]) p.noprev[idx]=p.no_f[idx];
    return;
  }
  idx -= NB*CN*32;
  if (idx < NB*HID){
    int b=(int)(idx/HID); int c=(int)(idx%HID);
    float o1 = p.anyf[b*2]   ? ord2f(p.uom1 [idx]) : 0.f;
    float o2 = p.anyf[b*2+1] ? ord2f(p.uom23[idx]) : 0.f;
    p.outs[((size_t)b*NT+t)*HID + c] = o1 + ((t>0)? o2 : 0.f);
  }
}

// ---------------- temporal attention pooling ----------------
__global__ void k_pool(Ptrs p){
  int b = blockIdx.x, tid = threadIdx.x;
  __shared__ float w[NT];
  if (tid < NT){
    const float* ot = p.outs + ((size_t)b*NT+tid)*HID;
    float acc=0.f;
    for(int j=0;j<32;j++){
      float a=p.bd[j];
      for(int c=0;c<HID;c++) a += ot[c]*p.Wd[c*32+j];
      acc += a*p.ctx[j];
    }
    w[tid] = (tid < p.lens[b]) ? acc : NEGF;
  }
  __syncthreads();
  if (tid==0){
    float mx=w[0];
    for(int i=1;i<NT;i++) mx=fmaxf(mx,w[i]);
    float s=0.f;
    for(int i=0;i<NT;i++){ w[i]=__expf(w[i]-mx); s+=w[i]; }
    for(int i=0;i<NT;i++) w[i]/=s;
  }
  __syncthreads();
  for(int c=tid;c<HID;c+=blockDim.x){
    float acc=0.f;
    for(int tt=0;tt<NT;tt++) acc += w[tt]*p.outs[((size_t)b*NT+tt)*HID+c];
    p.pooled[b*HID+c]=acc;
  }
}

// ---------------- head: feat concat, batchnorm over B, classifier ----------------
__global__ void k_feat(Ptrs p){
  int idx = blockIdx.x*blockDim.x+threadIdx.x;
  if (idx >= NB*CATSZ) return;
  int b=idx/CATSZ, f=idx%CATSZ;
  float v;
  if      (f < 64)  v = p.ug_g[p.user[b*3+0]*64 + f];
  else if (f < 128) v = p.ug_a[p.user[b*3+1]*64 + (f-64)];
  else if (f < 160) v = p.ug_c[p.user[b*3+2]*32 + (f-128)];
  else if (f < 416) v = p.pooled[b*HID + (f-160)];
  else              v = p.text[((size_t)b*NT + (p.lens[b]-1))*TEXTD + (f-416)];
  p.feat[idx]=v;
}

__global__ void k_bn(Ptrs p){
  int f = blockIdx.x*blockDim.x+threadIdx.x;
  if (f >= CATSZ) return;
  float mean=0.f;
  for(int b=0;b<NB;b++) mean += p.feat[b*CATSZ+f];
  mean /= (float)NB;
  float var=0.f;
  for(int b=0;b<NB;b++){ float d=p.feat[b*CATSZ+f]-mean; var+=d*d; }
  var /= (float)NB;
  float inv = rsqrtf(var+1e-5f);
  for(int b=0;b<NB;b++)
    p.yb[b*CATSZ+f] = (p.feat[b*CATSZ+f]-mean)*inv*p.gam[f] + p.bet[f];
}

__global__ void k_cls(Ptrs p){
  int idx = blockIdx.x*blockDim.x+threadIdx.x;
  if (idx >= NB*OUTSZ) return;
  int b=idx/OUTSZ, n=idx%OUTSZ;
  float acc = p.bcls[n];
  for(int k=0;k<CATSZ;k++) acc += p.yb[b*CATSZ+k]*p.Wcls[(size_t)k*OUTSZ+n];
  p.out[idx]=acc;
}

// ---------------- launch ----------------
extern "C" void kernel_launch(void* const* d_in, const int* in_sizes, int n_in,
                              void* d_out, int out_size, void* d_ws, size_t ws_size,
                              hipStream_t stream) {
  (void)in_sizes; (void)n_in; (void)out_size; (void)ws_size;
  Ptrs p;
  p.code_x   =(const float*)d_in[0];
  p.divided  =(const float*)d_in[1];
  p.neighbors=(const float*)d_in[2];
  p.lens     =(const int*)  d_in[3];
  p.user     =(const int*)  d_in[4];
  p.cate     =(const float*)d_in[5];
  p.text     =(const float*)d_in[6];
  // d_in[7] = admission_times (unused)
  p.c_emb   =(const float*)d_in[8];  p.n_emb   =(const float*)d_in[9];
  p.u_emb   =(const float*)d_in[10]; p.cate_emb=(const float*)d_in[11];
  p.adj     =(const float*)d_in[12]; p.cate_adj=(const float*)d_in[13];
  p.Wg=(const float*)d_in[14]; p.bg=(const float*)d_in[15];
  p.Wc=(const float*)d_in[16]; p.bc=(const float*)d_in[17];
  p.Wi=(const float*)d_in[18]; p.bi=(const float*)d_in[19];
  p.Wh=(const float*)d_in[20]; p.bh=(const float*)d_in[21];
  p.Wq=(const float*)d_in[22]; p.Wk=(const float*)d_in[23]; p.Wv=(const float*)d_in[24];
  p.Wd=(const float*)d_in[25]; p.bd=(const float*)d_in[26]; p.ctx=(const float*)d_in[27];
  p.ug_g=(const float*)d_in[28]; p.ug_a=(const float*)d_in[29]; p.ug_c=(const float*)d_in[30];
  p.gam=(const float*)d_in[31]; p.bet=(const float*)d_in[32];
  p.Wcls=(const float*)d_in[33]; p.bcls=(const float*)d_in[34];
  p.out=(float*)d_out;

  char* w=(char*)d_ws; size_t off=0;
  auto alloc=[&](size_t bytes)->char*{
    char* r=w+off; off=(off+bytes+255)&~(size_t)255; return r; };
  p.adjbf =(__bf16*)alloc((size_t)CN*CN*2);
  p.WgT   =(__bf16*)alloc(32*64*2);
  p.WiT   =(__bf16*)alloc(768*32*2);
  p.WhT   =(__bf16*)alloc(768*256*2);
  p.WqT   =(__bf16*)alloc(32*32*2);
  p.WkT   =(__bf16*)alloc(32*32*2);
  p.wvT   =(__bf16*)alloc(256*32*2);
  p.cadjbf=(__bf16*)alloc((size_t)CN*32*2);
  p.ceT   =(__bf16*)alloc((size_t)NB*CS*CN*2);
  p.neT   =(__bf16*)alloc((size_t)NB*CS*CN*2);
  p.tmp2T =(__bf16*)alloc((size_t)NB*32*32*2);
  p.Xc    =(__bf16*)alloc((size_t)NB*CN*64*2);
  p.Xn    =(__bf16*)alloc((size_t)NB*CN*64*2);
  p.cobf  =(__bf16*)alloc((size_t)NB*CN*32*2);
  p.xbf   =(__bf16*)alloc((size_t)NB*CN*32*2);
  p.Qh    =(__bf16*)alloc((size_t)NB*CN*32*2);
  p.Kh    =(__bf16*)alloc((size_t)NB*CN*32*2);
  p.VTm   =(__bf16*)alloc((size_t)NB*HID*CN*2);
  p.hbf   =(__bf16*)alloc((size_t)NB*CN*HID*2);
  p.m1v   =(float*)alloc((size_t)NB*CN*4);
  p.m2v   =(float*)alloc((size_t)NB*CN*4);
  p.m23v  =(float*)alloc((size_t)NB*CN*4);
  p.no_f  =(float*)alloc((size_t)NB*CN*32*4);
  p.noprev=(float*)alloc((size_t)NB*CN*32*4);
  p.h_f   =(float*)alloc((size_t)NB*CN*HID*4);
  p.hm1   =(float*)alloc((size_t)NB*CN*HID*4);
  p.outs  =(float*)alloc((size_t)NB*NT*HID*4);
  p.pooled=(float*)alloc((size_t)NB*HID*4);
  p.feat  =(float*)alloc((size_t)NB*CATSZ*4);
  p.yb    =(float*)alloc((size_t)NB*CATSZ*4);
  p.uom1  =(unsigned*)alloc((size_t)NB*HID*4);
  p.uom23 =(unsigned*)alloc((size_t)NB*HID*4);
  p.anyf  =(int*)alloc((size_t)NB*2*4);

  const long long TOT_INIT =
      (long long)CN*CN + 32*64 + 768*32 + 768*256 + 32*32 + 256*32 + CN*32 +
      (long long)NB*CN*HID + NB*CN*32 + (long long)NB*CN*64;
  const long long TOT_SPREP = (long long)NB*CS*CN + NB*CN + NB*32*32 + NB*HID;
  const long long TOT_CARRY = (long long)NB*CN*HID + NB*CN*32 + NB*HID;

  k_init<<<(unsigned)((TOT_INIT+255)/256),256,0,stream>>>(p);
  for (int t=0; t<NT; ++t){
    k_sprep<<<(unsigned)((TOT_SPREP+255)/256),256,0,stream>>>(p,t);
    k_adj  <<<NB*96,      32,0,stream>>>(p,t);
    k_co   <<<NB*96,      32,0,stream>>>(p);
    k_qk   <<<NB*96,      32,0,stream>>>(p);
    k_vt   <<<NB*16,      32,0,stream>>>(p);
    k_gru  <<<NB*96*16,   32,0,stream>>>(p);
    k_flash<<<NB*96,      32,0,stream>>>(p);
    k_any  <<<NB,        256,0,stream>>>(p);
    k_carry<<<(unsigned)((TOT_CARRY+255)/256),256,0,stream>>>(p,t);
  }
  k_pool<<<NB,256,0,stream>>>(p);
  k_feat<<<(NB*CATSZ+255)/256,256,0,stream>>>(p);
  k_bn  <<<(CATSZ+255)/256,256,0,stream>>>(p);
  k_cls <<<(NB*OUTSZ+255)/256,256,0,stream>>>(p);
}
